// MolGNN_GPS_pooling_73023033967228
// MI455X (gfx1250) — matrix operand order repaired and checked by reading, not compiled, not run on hardware
//
#include <hip/hip_runtime.h>
#include <hip/hip_bf16.h>

typedef __attribute__((ext_vector_type(16))) _Float16 v16h;
typedef __attribute__((ext_vector_type(2)))  _Float16 v2h;
typedef __attribute__((ext_vector_type(8)))  float    v8f;

#define NN   131072        // nodes
#define EE   262144        // directed edges
#define GG   4096          // graphs
#define MM   32            // nodes per graph
#define EPG  64            // edges per graph
#define HID  256
#define OUTC 768
#define BN_EPS 1e-5f

// ---------------------------------------------------------------------------
// WMMA GEMM: C[M,N] = act(A[M,K]) * B + bias ;  B is [K,N] or [N,K] (TRANS_B)
// block = 256 threads (8 wave32), block tile 128x128, k-chunk 32.
// LDS tiles stored fragment-major so each lane's WMMA operand is one
// contiguous 32B v16h (2x ds_load_b128, no repacking moves).
// Double-buffered: next chunk loaded float2-wide into registers while
// WMMAs consume the current buffer.
// ---------------------------------------------------------------------------
#define BM 128
#define BNT 128
#define BK 32
#define LDW 48   // padded row stride (halfs): 96B, keeps 32B-aligned fragments

template <bool TRANS_B, bool RELU_A>
__global__ __launch_bounds__(256)
void gemm_wmma(const float* __restrict__ A, const float* __restrict__ B,
               const float* __restrict__ bias, float* __restrict__ C,
               int Mrows, int Ncols, int K)
{
    (void)Mrows;
    // sA[buf][row][perm(k)]   perm swaps bit3<->bit4 so a fragment is contiguous
    // sBT[buf][col][perm(k)]  transposed B tile; same bit3<->bit4 swap as sA
    __shared__ __attribute__((aligned(32))) _Float16 sA[2][BM][LDW];
    __shared__ __attribute__((aligned(32))) _Float16 sBT[2][BNT][LDW];

    const int tid  = threadIdx.x;
    const int lane = tid & 31;
    const int w    = tid >> 5;    // 0..7
    const int wm   = w >> 1;      // 0..3 -> 32-row slab
    const int wn   = w & 1;       // 0..1 -> 64-col slab
    const int half = lane >> 4;   // lane half (ISA A/B/C layouts)
    const int l15  = lane & 15;
    const int row0 = blockIdx.y * BM;
    const int col0 = blockIdx.x * BNT;

    v8f acc[2][4];
#pragma unroll
    for (int i = 0; i < 2; ++i)
#pragma unroll
        for (int j = 0; j < 4; ++j)
            acc[i][j] = (v8f){0.f,0.f,0.f,0.f,0.f,0.f,0.f,0.f};

    // per-thread staging assignments (8 float2 each for A and B)
    const int ar = tid >> 4;            // base row (A) / col (B), +16 per iter... no:
    // idx = it*256 + tid ; r = idx>>4 ; pair = idx&15
    float2 ra[8], rb[8];

    auto load_regs = [&](int kk) {
#pragma unroll
        for (int it = 0; it < 8; ++it) {
            int idx = it * 256 + tid;
            int r = idx >> 4, c = (idx & 15) * 2;
            ra[it] = *(const float2*)&A[(size_t)(row0 + r) * K + kk + c];
        }
#pragma unroll
        for (int it = 0; it < 8; ++it) {
            int idx = it * 256 + tid;
            int n = idx >> 4, kc = (idx & 15) * 2;
            if (TRANS_B) {
                rb[it] = *(const float2*)&B[(size_t)(col0 + n) * K + kk + kc];
            } else {
                rb[it].x = B[(size_t)(kk + kc)     * Ncols + col0 + n];
                rb[it].y = B[(size_t)(kk + kc + 1) * Ncols + col0 + n];
            }
        }
    };
    auto store_lds = [&](int buf) {
#pragma unroll
        for (int it = 0; it < 8; ++it) {
            int idx = it * 256 + tid;
            int r = idx >> 4, c = (idx & 15) * 2;
            float x = ra[it].x, y = ra[it].y;
            if (RELU_A) { x = x > 0.f ? x : 0.f; y = y > 0.f ? y : 0.f; }
            int p = (c & 7) | ((c & 8) << 1) | ((c & 16) >> 1);  // bit3<->bit4
            v2h pv = { (_Float16)x, (_Float16)y };
            *(v2h*)&sA[buf][r][p] = pv;
        }
#pragma unroll
        for (int it = 0; it < 8; ++it) {
            int idx = it * 256 + tid;
            int n = idx >> 4, kc = (idx & 15) * 2;
            v2h pv = { (_Float16)rb[it].x, (_Float16)rb[it].y };
            *(v2h*)&sBT[buf][n][(kc & 7) | ((kc & 8) << 1) | ((kc & 16) >> 1)] = pv;
        }
    };

    // prologue: chunk 0 -> buffer 0
    load_regs(0);
    store_lds(0);
    __syncthreads();

    int cur = 0;
    for (int k0 = 0; k0 < K; k0 += BK) {
        const bool has_next = (k0 + BK) < K;
        if (has_next) load_regs(k0 + BK);   // HBM -> regs, overlaps WMMA below

        // fragments: one contiguous 32B v16h per operand (2x ds_load_b128)
        v16h af[2];
#pragma unroll
        for (int tm = 0; tm < 2; ++tm) {
            int m = wm * 32 + tm * 16 + l15;
            af[tm] = *(const v16h*)&sA[cur][m][16 * half];
        }
        v16h bf[4];
#pragma unroll
        for (int tn = 0; tn < 4; ++tn) {
            int n = wn * 64 + tn * 16 + l15;
            bf[tn] = *(const v16h*)&sBT[cur][n][16 * half];
        }
#pragma unroll
        for (int tm = 0; tm < 2; ++tm)
#pragma unroll
            for (int tn = 0; tn < 4; ++tn)
                acc[tm][tn] = __builtin_amdgcn_wmma_f32_16x16x32_f16(
                    false, af[tm], false, bf[tn], (short)0, acc[tm][tn],
                    false, false);

        if (has_next) store_lds(cur ^ 1);   // regs -> other buffer
        __syncthreads();
        cur ^= 1;
    }

    // epilogue: C/D layout — VGPR r holds M = r + 8*half, N = lane&15
#pragma unroll
    for (int tm = 0; tm < 2; ++tm)
#pragma unroll
        for (int tn = 0; tn < 4; ++tn) {
            int col = col0 + wn * 64 + tn * 16 + l15;
            float bv = bias ? bias[col] : 0.f;
#pragma unroll
            for (int r = 0; r < 8; ++r) {
                int row = row0 + wm * 32 + tm * 16 + half * 8 + r;
                C[(size_t)row * Ncols + col] = acc[tm][tn][r] + bv;
            }
        }
}

// ---------------------------------------------------------------------------
// Embedding sums
// ---------------------------------------------------------------------------
__global__ __launch_bounds__(256)
void embed_nodes(const int* __restrict__ x,
                 const float* t0, const float* t1, const float* t2,
                 const float* t3, const float* t4, const float* t5,
                 const float* t6, const float* t7, const float* t8,
                 float* __restrict__ h)
{
    int n = blockIdx.x, c = threadIdx.x;
    const int* xr = x + (size_t)n * 9;
    float s = t0[(size_t)xr[0] * HID + c] + t1[(size_t)xr[1] * HID + c]
            + t2[(size_t)xr[2] * HID + c] + t3[(size_t)xr[3] * HID + c]
            + t4[(size_t)xr[4] * HID + c] + t5[(size_t)xr[5] * HID + c]
            + t6[(size_t)xr[6] * HID + c] + t7[(size_t)xr[7] * HID + c]
            + t8[(size_t)xr[8] * HID + c];
    h[(size_t)n * HID + c] = s;
}

__global__ __launch_bounds__(256)
void embed_edges(const int* __restrict__ ea,
                 const float* t0, const float* t1, const float* t2,
                 float* __restrict__ e)
{
    int n = blockIdx.x, c = threadIdx.x;
    const int* er = ea + (size_t)n * 3;
    e[(size_t)n * HID + c] = t0[(size_t)er[0] * HID + c]
                           + t1[(size_t)er[1] * HID + c]
                           + t2[(size_t)er[2] * HID + c];
}

// ---------------------------------------------------------------------------
// GINE aggregate: one block per graph, LDS scatter (edges are graph-local)
// z = (1+eps)*h + segment_sum(relu(h[src] + e), dst)
// ---------------------------------------------------------------------------
__global__ __launch_bounds__(256)
void gine_agg(const float* __restrict__ h, const float* __restrict__ e,
              const int* __restrict__ src, const int* __restrict__ dst,
              const float* __restrict__ epsp, float* __restrict__ z)
{
    __shared__ float acc[MM * HID];   // 32 KB, thread t owns channel t
    const int g = blockIdx.x, t = threadIdx.x;
#pragma unroll
    for (int r = 0; r < MM; ++r) acc[r * HID + t] = 0.f;
    const int e0 = g * EPG;
    for (int ei = 0; ei < EPG; ++ei) {
        int s = src[e0 + ei];
        int d = dst[e0 + ei] - g * MM;
        float m = h[(size_t)s * HID + t] + e[(size_t)(e0 + ei) * HID + t];
        acc[d * HID + t] += (m > 0.f ? m : 0.f);
    }
    float k = 1.0f + epsp[0];
#pragma unroll
    for (int r = 0; r < MM; ++r) {
        size_t n = (size_t)(g * MM + r) * HID + t;
        z[n] = k * h[n] + acc[r * HID + t];
    }
}

// ---------------------------------------------------------------------------
// BatchNorm (training-mode, biased var): fused add + stats, then apply
// ---------------------------------------------------------------------------
__global__ __launch_bounds__(256)
void zero_f32(float* p, int n)
{
    int i = blockIdx.x * 256 + threadIdx.x;
    if (i < n) p[i] = 0.f;
}

__global__ __launch_bounds__(256)
void add_stats(const float* __restrict__ A, const float* __restrict__ B,
               float* __restrict__ X, float* __restrict__ stats, int C)
{
    const int RPB = 1024;
    int r0 = blockIdx.x * RPB;
    for (int c = threadIdx.x; c < C; c += 256) {
        float s = 0.f, s2 = 0.f;
        for (int r = r0; r < r0 + RPB; ++r) {
            size_t idx = (size_t)r * C + c;
            float v = A[idx] + (B ? B[idx] : 0.f);
            X[idx] = v;
            s += v; s2 += v * v;
        }
        atomicAdd(&stats[c], s);
        atomicAdd(&stats[C + c], s2);
    }
}

template <bool RELU>
__global__ __launch_bounds__(256)
void bn_apply(const float* __restrict__ X, const float* __restrict__ stats,
              const float* __restrict__ gam, const float* __restrict__ bet,
              float* __restrict__ Y, int Nr, int C)
{
    size_t i = (size_t)blockIdx.x * 256 + threadIdx.x;
    if (i >= (size_t)Nr * C) return;
    int c = (int)(i & (size_t)(C - 1));
    float invn = 1.0f / (float)Nr;
    float mu  = stats[c] * invn;
    float var = stats[C + c] * invn - mu * mu;
    float v = (X[i] - mu) * rsqrtf(var + BN_EPS) * gam[c] + bet[c];
    if (RELU) v = v > 0.f ? v : 0.f;
    Y[i] = v;
}

__global__ __launch_bounds__(256)
void add_ew(const float* __restrict__ a, const float* __restrict__ b,
            float* __restrict__ c, size_t n)
{
    size_t i = (size_t)blockIdx.x * 256 + threadIdx.x;
    if (i < n) c[i] = a[i] + b[i];
}

// ---------------------------------------------------------------------------
// Per-(graph,head) attention: q,k,v = [32,64]; scores/softmax/AV in LDS
// ---------------------------------------------------------------------------
__global__ __launch_bounds__(256)
void attention(const float* __restrict__ qkv, float* __restrict__ o)
{
    __shared__ float sq[MM * 64], sk[MM * 64], sv[MM * 64], ss[MM * 33];
    const int g = blockIdx.x >> 2, hh = blockIdx.x & 3, t = threadIdx.x;
    for (int idx = t; idx < MM * 64; idx += 256) {
        int i = idx >> 6, d = idx & 63;
        size_t base = (size_t)(g * MM + i) * (3 * HID) + hh * 64 + d;
        sq[idx] = qkv[base];
        sk[idx] = qkv[base + HID];
        sv[idx] = qkv[base + 2 * HID];
    }
    __syncthreads();
    for (int p = t; p < MM * MM; p += 256) {
        int i = p >> 5, j = p & 31;
        float s = 0.f;
        for (int d = 0; d < 64; ++d) s += sq[i * 64 + d] * sk[j * 64 + d];
        ss[i * 33 + j] = s * 0.125f;   // 1/sqrt(64)
    }
    __syncthreads();
    if (t < MM) {
        float mx = -1e30f;
        for (int j = 0; j < MM; ++j) mx = fmaxf(mx, ss[t * 33 + j]);
        float sum = 0.f;
        for (int j = 0; j < MM; ++j) {
            float ev = __expf(ss[t * 33 + j] - mx);
            ss[t * 33 + j] = ev; sum += ev;
        }
        float inv = 1.f / sum;
        for (int j = 0; j < MM; ++j) ss[t * 33 + j] *= inv;
    }
    __syncthreads();
    for (int p = t; p < MM * 64; p += 256) {
        int i = p >> 6, d = p & 63;
        float s = 0.f;
        for (int j = 0; j < MM; ++j) s += ss[i * 33 + j] * sv[j * 64 + d];
        o[(size_t)(g * MM + i) * HID + hh * 64 + d] = s;
    }
}

// ---------------------------------------------------------------------------
// Pool (batch is contiguous 32-node groups) and L2 normalize (in place)
// ---------------------------------------------------------------------------
__global__ __launch_bounds__(256)
void pool32(const float* __restrict__ h, float* __restrict__ gv)
{
    int g = blockIdx.x, t = threadIdx.x;
    float s = 0.f;
#pragma unroll
    for (int r = 0; r < MM; ++r) s += h[(size_t)(g * MM + r) * HID + t];
    gv[(size_t)g * HID + t] = s;
}

__global__ __launch_bounds__(256)
void l2norm_rows(float* __restrict__ X, int C)
{
    __shared__ float red[256];
    int row = blockIdx.x, t = threadIdx.x;
    float s = 0.f;
    for (int c = t; c < C; c += 256) {
        float v = X[(size_t)row * C + c];
        s += v * v;
    }
    red[t] = s; __syncthreads();
    for (int st = 128; st > 0; st >>= 1) {
        if (t < st) red[t] += red[t + st];
        __syncthreads();
    }
    float inv = 1.f / fmaxf(sqrtf(red[0]), 1e-12f);
    for (int c = t; c < C; c += 256) X[(size_t)row * C + c] *= inv;
}

// ---------------------------------------------------------------------------
// Host orchestration
// ---------------------------------------------------------------------------
// Input flattening (setup_inputs insertion order, params recursed first):
//  0..8  atom_emb[i]         9..11 bond_emb[i]
//  12 + 23*l + off : per-layer params (offsets below)
//  104 proj_w1, 105 proj_b1, 106 proj_w2, 107 proj_b2
//  108 x, 109 edge_attr, 110 edge_index, 111 batch
enum { P_EPS = 0, P_LIN_W, P_LIN_B, P_NN1_W, P_NN1_B, P_NNBN_G, P_NNBN_B,
       P_NN2_W, P_NN2_B, P_BN1_G, P_BN1_B, P_IN_W, P_IN_B, P_OUT_W, P_OUT_B,
       P_BN2_G, P_BN2_B, P_W1, P_B1, P_W2, P_B2, P_BN3_G, P_BN3_B };
#define LP(l, off) ((const float*)d_in[12 + 23 * (l) + (off)])

extern "C" void kernel_launch(void* const* d_in, const int* in_sizes, int n_in,
                              void* d_out, int out_size, void* d_ws, size_t ws_size,
                              hipStream_t stream)
{
    (void)in_sizes; (void)n_in; (void)out_size; (void)ws_size;

    const int* x_idx  = (const int*)d_in[108];
    const int* eattr  = (const int*)d_in[109];
    const int* src    = (const int*)d_in[110];          // edge_index[0]
    const int* dst    = src + EE;                       // edge_index[1]

    // workspace carve-out (floats)
    float* wp = (float*)d_ws;
    auto bump = [&](size_t nfloats) { float* p = wp; wp += nfloats; return p; };
    float* ea    = bump((size_t)EE * HID);      // bond embeddings
    float* bufE  = bump((size_t)EE * HID);      // projected edge features
    float* h     = bump((size_t)NN * HID);
    float* zbuf  = bump((size_t)NN * HID);
    float* A1    = bump((size_t)NN * 2 * HID);
    float* B1    = bump((size_t)NN * 2 * HID);
    float* qkv   = bump((size_t)NN * 3 * HID);
    float* obuf  = bump((size_t)NN * HID);
    float* hl    = bump((size_t)NN * HID);
    float* ha    = bump((size_t)NN * HID);
    float* outb  = bump((size_t)NN * HID);
    float* t1    = bump((size_t)NN * HID);
    float* t2    = bump((size_t)NN * HID);
    float* xb    = bump((size_t)NN * HID);
    float* stats = bump(1024);
    float* gvec  = bump((size_t)GG * HID);
    float* gmid  = bump((size_t)GG * HID);

    auto gemm = [&](const float* A, const float* B, const float* bias, float* C,
                    int M_, int N_, int K_, bool transB, bool reluA) {
        dim3 grid(N_ / BNT, M_ / BM);
        if (transB) {
            if (reluA) gemm_wmma<true,  true ><<<grid, 256, 0, stream>>>(A, B, bias, C, M_, N_, K_);
            else       gemm_wmma<true,  false><<<grid, 256, 0, stream>>>(A, B, bias, C, M_, N_, K_);
        } else {
            if (reluA) gemm_wmma<false, true ><<<grid, 256, 0, stream>>>(A, B, bias, C, M_, N_, K_);
            else       gemm_wmma<false, false><<<grid, 256, 0, stream>>>(A, B, bias, C, M_, N_, K_);
        }
    };
    auto bn = [&](const float* A, const float* B, float* X, float* Y,
                  const float* g_, const float* b_, int C, bool relu) {
        zero_f32<<<4, 256, 0, stream>>>(stats, 1024);
        add_stats<<<NN / 1024, 256, 0, stream>>>(A, B, X, stats, C);
        int blocks = (int)(((size_t)NN * C + 255) / 256);
        if (relu) bn_apply<true ><<<blocks, 256, 0, stream>>>(X, stats, g_, b_, Y, NN, C);
        else      bn_apply<false><<<blocks, 256, 0, stream>>>(X, stats, g_, b_, Y, NN, C);
    };

    // embeddings
    embed_nodes<<<NN, HID, 0, stream>>>(x_idx,
        (const float*)d_in[0], (const float*)d_in[1], (const float*)d_in[2],
        (const float*)d_in[3], (const float*)d_in[4], (const float*)d_in[5],
        (const float*)d_in[6], (const float*)d_in[7], (const float*)d_in[8], h);
    embed_edges<<<EE, HID, 0, stream>>>(eattr,
        (const float*)d_in[9], (const float*)d_in[10], (const float*)d_in[11], ea);

    for (int l = 0; l < 4; ++l) {
        // GINEConv
        gemm(ea, LP(l, P_LIN_W), LP(l, P_LIN_B), bufE, EE, HID, HID, false, false);
        gine_agg<<<GG, HID, 0, stream>>>(h, bufE, src, dst, LP(l, P_EPS), zbuf);
        gemm(zbuf, LP(l, P_NN1_W), LP(l, P_NN1_B), A1, NN, 2 * HID, HID, false, false);
        bn(A1, nullptr, A1, B1, LP(l, P_NNBN_G), LP(l, P_NNBN_B), 2 * HID, /*relu*/true);
        gemm(B1, LP(l, P_NN2_W), LP(l, P_NN2_B), t1, NN, HID, 2 * HID, false, false);
        bn(t1, h, xb, hl, LP(l, P_BN1_G), LP(l, P_BN1_B), HID, false);

        // global multihead attention
        gemm(h, LP(l, P_IN_W), LP(l, P_IN_B), qkv, NN, 3 * HID, HID, /*transB*/true, false);
        attention<<<GG * 4, 256, 0, stream>>>(qkv, obuf);
        gemm(obuf, LP(l, P_OUT_W), LP(l, P_OUT_B), t2, NN, HID, HID, /*transB*/true, false);
        bn(t2, h, xb, ha, LP(l, P_BN2_G), LP(l, P_BN2_B), HID, false);

        // combine + GPS MLP
        add_ew<<<(int)(((size_t)NN * HID + 255) / 256), 256, 0, stream>>>(hl, ha, outb, (size_t)NN * HID);
        gemm(outb, LP(l, P_W1), LP(l, P_B1), A1, NN, 2 * HID, HID, false, false);
        gemm(A1, LP(l, P_W2), LP(l, P_B2), t1, NN, HID, 2 * HID, false, /*reluA*/true);
        bn(outb, t1, xb, h, LP(l, P_BN3_G), LP(l, P_BN3_B), HID, false);
    }

    // pool + projection + L2 normalize
    pool32<<<GG, HID, 0, stream>>>(h, gvec);
    gemm(gvec, (const float*)d_in[104], (const float*)d_in[105], gmid, GG, HID, HID, false, false);
    gemm(gmid, (const float*)d_in[106], (const float*)d_in[107], (float*)d_out,
         GG, OUTC, HID, false, /*reluA*/true);
    l2norm_rows<<<GG, 256, 0, stream>>>((float*)d_out, OUTC);
}
